// DynamicRNN_45561013076397
// MI455X (gfx1250) — compile-verified, matrix operand order checked by reading, not run on hardware
//
#include <hip/hip_runtime.h>
#include <hip/hip_bf16.h>

typedef __attribute__((ext_vector_type(16))) _Float16 v16h;
typedef __attribute__((ext_vector_type(8)))  float    v8f;

#define T_SEQ 128

#if __has_builtin(__builtin_amdgcn_global_load_async_to_lds_b128) && \
    __has_builtin(__builtin_amdgcn_s_wait_asynccnt)
#define USE_ASYNC_LDS 1
#else
#define USE_ASYNC_LDS 0
#endif

// ---------------- fast activations (v_exp_f32 / v_rcp_f32 TRANS path) ----------
__device__ __forceinline__ float sigm(float x) {
  return __builtin_amdgcn_rcpf(1.0f + __expf(-x));
}
__device__ __forceinline__ float tanh_fast(float x) {
  return 2.0f * __builtin_amdgcn_rcpf(1.0f + __expf(-2.0f * x)) - 1.0f;
}

// ---------------- one LSTM cell step, H=5, In=5, weights broadcast from LDS ----
__device__ __forceinline__ void lstm_cell(const float* xin,
    const float* __restrict__ wih, const float* __restrict__ whh,
    const float* __restrict__ bias, float* h, float* c)
{
  float g[20];
#pragma unroll
  for (int n = 0; n < 20; ++n) {
    float a = bias[n];
#pragma unroll
    for (int k = 0; k < 5; ++k) a = fmaf(xin[k], wih[n * 5 + k], a);
#pragma unroll
    for (int j = 0; j < 5; ++j) a = fmaf(h[j], whh[n * 5 + j], a);
    g[n] = a;
  }
  float hn[5];
#pragma unroll
  for (int j = 0; j < 5; ++j) {           // PyTorch gate order i,f,g,o
    float i_ = sigm(g[j]);
    float f_ = sigm(g[5 + j]);
    float g_ = tanh_fast(g[10 + j]);
    float o_ = sigm(g[15 + j]);
    c[j]  = f_ * c[j] + i_ * g_;
    hn[j] = o_ * tanh_fast(c[j]);
  }
#pragma unroll
  for (int j = 0; j < 5; ++j) h[j] = hn[j];
}

#if USE_ASYNC_LDS
// Builtin signature (from hipcc diagnostic): param0 = AS1 pointer to int32x4
// (vector_size form, non-const); param1 = LDS-side pointer; then imm offset, cpol.
typedef int async_v4i __attribute__((vector_size(4 * sizeof(int))));
typedef __attribute__((address_space(1))) async_v4i* g_v4i_p;
typedef __attribute__((address_space(3))) async_v4i* l_v4i_p;

// one 16B async global->LDS copy (ASYNCcnt-tracked, no VGPR round trip)
__device__ __forceinline__ void async_cp16(const float* g, float* l) {
  __builtin_amdgcn_global_load_async_to_lds_b128((g_v4i_p)g, (l_v4i_p)l, 0, 0);
}
#endif

// ---------------- kernel 1: recurrent part. role 0 = lstm1(x1); role 1 = lstm2a+2b(x2)
__global__ __launch_bounds__(256) void lstm_kernel(
    const float* __restrict__ x1, const float* __restrict__ x2,
    const float* __restrict__ Wih1,  const float* __restrict__ Whh1,
    const float* __restrict__ bih1,  const float* __restrict__ bhh1,
    const float* __restrict__ Wih2a, const float* __restrict__ Whh2a,
    const float* __restrict__ bih2a, const float* __restrict__ bhh2a,
    const float* __restrict__ Wih2b, const float* __restrict__ Whh2b,
    const float* __restrict__ bih2b, const float* __restrict__ bhh2b,
    float* __restrict__ out1, float* __restrict__ out2, int B)
{
  __shared__ float wa_ih[100], wa_hh[100], wa_b[20];
  __shared__ float wb_ih[100], wb_hh[100], wb_b[20];
#if USE_ASYNC_LDS
  __shared__ float xstage[2][256 * 20];   // double buffer: 80 B per lane per chunk
#endif
  const int role = blockIdx.y;   // uniform per block
  const int tid  = threadIdx.x;

  if (role == 0) {
    for (int i = tid; i < 100; i += 256) { wa_ih[i] = Wih1[i]; wa_hh[i] = Whh1[i]; }
    if (tid < 20) wa_b[tid] = bih1[tid] + bhh1[tid];
  } else {
    for (int i = tid; i < 100; i += 256) {
      wa_ih[i] = Wih2a[i]; wa_hh[i] = Whh2a[i];
      wb_ih[i] = Wih2b[i]; wb_hh[i] = Whh2b[i];
    }
    if (tid < 20) { wa_b[tid] = bih2a[tid] + bhh2a[tid];
                    wb_b[tid] = bih2b[tid] + bhh2b[tid]; }
  }
  __syncthreads();

  const int b = blockIdx.x * 256 + tid;
  if (b >= B) return;

  const float* xp = (role == 0 ? x1 : x2) + (size_t)b * T_SEQ * 5;
  float hA[5] = {}, cA[5] = {}, hB[5] = {}, cB[5] = {};

#if USE_ASYNC_LDS
  float* st0 = &xstage[0][tid * 20];
  float* st1 = &xstage[1][tid * 20];
  // prologue: async-stage chunk 0 (4 timesteps = 20 floats = 5 x b128)
#pragma unroll
  for (int q = 0; q < 5; ++q) async_cp16(xp + q * 4, st0 + q * 4);

  for (int t0 = 0; t0 < T_SEQ; t0 += 4) {
    const int i = t0 >> 2;
    float* cur = (i & 1) ? st1 : st0;
    float* nxt = (i & 1) ? st0 : st1;
    if (t0 + 4 < T_SEQ) {
      // overlap: issue next chunk while computing this one
#pragma unroll
      for (int q = 0; q < 5; ++q) async_cp16(xp + (t0 + 4) * 5 + q * 4, nxt + q * 4);
      __builtin_amdgcn_s_wait_asynccnt(5);   // in-order: current chunk has landed
    } else {
      __builtin_amdgcn_s_wait_asynccnt(0);
    }
    float xs[20];
#pragma unroll
    for (int q = 0; q < 20; ++q) xs[q] = cur[q];   // ds_load_b128 x5
#pragma unroll
    for (int s = 0; s < 4; ++s) {
      lstm_cell(&xs[s * 5], wa_ih, wa_hh, wa_b, hA, cA);
      if (role == 1) lstm_cell(hA, wb_ih, wb_hh, wb_b, hB, cB);
    }
  }
#else
  for (int t0 = 0; t0 < T_SEQ; t0 += 4) {
    if (t0 + 8 < T_SEQ) __builtin_prefetch(xp + (t0 + 8) * 5, 0, 3); // global_prefetch_b8
    float xs[20];
    const float4* p = (const float4*)(xp + t0 * 5);
#pragma unroll
    for (int q = 0; q < 5; ++q) {
      float4 v = p[q];
      xs[4*q+0] = v.x; xs[4*q+1] = v.y; xs[4*q+2] = v.z; xs[4*q+3] = v.w;
    }
#pragma unroll
    for (int s = 0; s < 4; ++s) {
      lstm_cell(&xs[s * 5], wa_ih, wa_hh, wa_b, hA, cA);
      if (role == 1) lstm_cell(hA, wb_ih, wb_hh, wb_b, hB, cB);
    }
  }
#endif

  if (role == 0) {
#pragma unroll
    for (int j = 0; j < 5; ++j) out1[b * 5 + j] = hA[j];
  } else {
#pragma unroll
    for (int j = 0; j < 5; ++j) out2[b * 5 + j] = hB[j];
  }
}

// ---------------- WMMA fragment builders (16-bit A layout from CDNA5 ISA)
__device__ __forceinline__ v16h load_afrag(const _Float16* act, int lane) {
  const int m = lane & 15, hi = (lane >> 4) & 1;
  v16h a;
#pragma unroll
  for (int e = 0; e < 16; ++e) {
    const int k = (e < 8 ? e : e + 8) + hi * 8;   // K swizzle per ISA layout
    a[e] = act[m * 32 + k];
  }
  return a;
}

template<int K, int N>
__device__ __forceinline__ v16h load_bfrag(const _Float16* w, int ntile, int lane) {
  const int n = (lane & 15) + ntile * 16, hi = (lane >> 4) & 1;
  v16h bf;
#pragma unroll
  for (int e = 0; e < 16; ++e) {
    const int k = (e < 8 ? e : e + 8) + hi * 8;
    bf[e] = (k < K && n < N) ? w[n * K + k] : (_Float16)0.0f;  // zero-pad K,N
  }
  return bf;
}

// One FC layer for a 16-row tile: dst[16,N] = ReLU(src[16,K] * W^T + b), fp32 acc
template<int K, int N, bool RELU>
__device__ __forceinline__ void fc_layer(const _Float16* src, _Float16* dst,
    const _Float16* w, const float* bias, int lane)
{
  const v16h a  = load_afrag(src, lane);
  const int  hi = (lane >> 4) & 1;
#pragma unroll
  for (int nt = 0; nt < (N + 15) / 16; ++nt) {
    const v16h bf  = load_bfrag<K, N>(w, nt, lane);
    const int  col = (lane & 15) + nt * 16;
    const float bv = (col < N) ? bias[col] : 0.0f;   // bias folded into C
    v8f acc;
#pragma unroll
    for (int r = 0; r < 8; ++r) acc[r] = bv;
    acc = __builtin_amdgcn_wmma_f32_16x16x32_f16(false, a, false, bf,
                                                 (short)0, acc, false, false);
#pragma unroll
    for (int r = 0; r < 8; ++r) {                    // C/D: row=r+8*hi, col=lane%16
      float v = acc[r];
      if (RELU && v < 0.0f) v = 0.0f;
      dst[(r + 8 * hi) * 32 + col] = (_Float16)v;
    }
  }
  if (N < 32 && lane < 16) {                         // keep K-padding NaN-free
#pragma unroll
    for (int k = N; k < 32; ++k) dst[lane * 32 + k] = (_Float16)0.0f;
  }
}

// ---------------- kernel 2: FC head. 4 waves/block, one 16-row batch tile per wave
__global__ __launch_bounds__(128) void fc_wmma_kernel(
    const float* __restrict__ x1, const float* __restrict__ x2,
    const float* __restrict__ out1, const float* __restrict__ out2,
    const float* __restrict__ W1, const float* __restrict__ b1,
    const float* __restrict__ W2, const float* __restrict__ b2,
    const float* __restrict__ W3, const float* __restrict__ b3,
    const float* __restrict__ W4, const float* __restrict__ b4,
    const float* __restrict__ W5, const float* __restrict__ b5,
    float* __restrict__ out, int B)
{
  __shared__ _Float16 w1h[32 * 20];
  __shared__ _Float16 w2h[32 * 32];
  __shared__ _Float16 w3h[16 * 32];
  __shared__ _Float16 w4h[16 * 16];
  __shared__ _Float16 w5h[5 * 16];
  __shared__ float    bsh[101];
  __shared__ _Float16 actA[4][16 * 32];
  __shared__ _Float16 actB[4][16 * 32];

  const int tid = threadIdx.x;
  for (int i = tid; i < 32 * 20; i += 128) w1h[i] = (_Float16)W1[i];
  for (int i = tid; i < 32 * 32; i += 128) w2h[i] = (_Float16)W2[i];
  for (int i = tid; i < 16 * 32; i += 128) w3h[i] = (_Float16)W3[i];
  for (int i = tid; i < 16 * 16; i += 128) w4h[i] = (_Float16)W4[i];
  for (int i = tid; i < 5 * 16;  i += 128) w5h[i] = (_Float16)W5[i];
  if      (tid < 32)  bsh[tid] = b1[tid];
  else if (tid < 64)  bsh[tid] = b2[tid - 32];
  else if (tid < 80)  bsh[tid] = b3[tid - 64];
  else if (tid < 96)  bsh[tid] = b4[tid - 80];
  else if (tid < 101) bsh[tid] = b5[tid - 96];
  __syncthreads();

  const int wave = tid >> 5, lane = tid & 31;
  const int ntiles = (B + 15) / 16;
  int tile = blockIdx.x * 4 + wave;
  if (tile >= ntiles) tile = ntiles - 1;   // clamp -> uniform barriers, EXEC all-1s at WMMA
  _Float16* A0 = actA[wave];
  _Float16* A1 = actB[wave];

  // Stage features [16,20] -> fp16, zero-pad K to 32. Lanes 0-15 own one row each.
  if (lane < 16) {
    const int m = lane;
    int bb = tile * 16 + m; if (bb >= B) bb = B - 1;
    const size_t last = (size_t)bb * T_SEQ * 5 + (size_t)(T_SEQ - 1) * 5;
#pragma unroll
    for (int k = 0; k < 5; ++k) {
      A0[m * 32 + 0  + k] = (_Float16)x1[last + k];
      A0[m * 32 + 5  + k] = (_Float16)x2[last + k];
      A0[m * 32 + 10 + k] = (_Float16)out1[bb * 5 + k];
      A0[m * 32 + 15 + k] = (_Float16)out2[bb * 5 + k];
    }
#pragma unroll
    for (int k = 20; k < 32; ++k) A0[m * 32 + k] = (_Float16)0.0f;
  }
  __syncthreads();

  fc_layer<20, 32, true>(A0, A1, w1h, bsh +  0, lane); __syncthreads();
  fc_layer<32, 32, true>(A1, A0, w2h, bsh + 32, lane); __syncthreads();
  fc_layer<32, 16, true>(A0, A1, w3h, bsh + 64, lane); __syncthreads();
  fc_layer<16, 16, true>(A1, A0, w4h, bsh + 80, lane); __syncthreads();

  // Final layer: [16,16] x (16->5), fp32 result straight to global
  {
    const v16h a  = load_afrag(A0, lane);
    const v16h bf = load_bfrag<16, 5>(w5h, 0, lane);
    const int  col = lane & 15, hi = (lane >> 4) & 1;
    const float bv = (col < 5) ? bsh[96 + col] : 0.0f;
    v8f acc;
#pragma unroll
    for (int r = 0; r < 8; ++r) acc[r] = bv;
    acc = __builtin_amdgcn_wmma_f32_16x16x32_f16(false, a, false, bf,
                                                 (short)0, acc, false, false);
    if (col < 5) {
#pragma unroll
      for (int r = 0; r < 8; ++r) {
        const int bb = tile * 16 + (r + 8 * hi);
        if (bb < B) out[bb * 5 + col] = acc[r];
      }
    }
  }
}

// ---------------- host launcher ------------------------------------------------
extern "C" void kernel_launch(void* const* d_in, const int* in_sizes, int n_in,
                              void* d_out, int out_size, void* d_ws, size_t ws_size,
                              hipStream_t stream) {
  const float* x1 = (const float*)d_in[0];
  const float* x2 = (const float*)d_in[1];
  const int B = in_sizes[0] / (T_SEQ * 5);

  float* out1 = (float*)d_ws;                   // [B,5]
  float* out2 = out1 + (size_t)B * 5;           // [B,5]
  float* out  = (float*)d_out;                  // [B,5]

  dim3 g1((B + 255) / 256, 2);                  // y=0: lstm1, y=1: lstm2a+2b
  lstm_kernel<<<g1, 256, 0, stream>>>(x1, x2,
      (const float*)d_in[2],  (const float*)d_in[3],  (const float*)d_in[4],  (const float*)d_in[5],
      (const float*)d_in[6],  (const float*)d_in[7],  (const float*)d_in[8],  (const float*)d_in[9],
      (const float*)d_in[10], (const float*)d_in[11], (const float*)d_in[12], (const float*)d_in[13],
      out1, out2, B);

  const int tiles = (B + 15) / 16;
  fc_wmma_kernel<<<(tiles + 3) / 4, 128, 0, stream>>>(x1, x2, out1, out2,
      (const float*)d_in[14], (const float*)d_in[15], (const float*)d_in[16], (const float*)d_in[17],
      (const float*)d_in[18], (const float*)d_in[19], (const float*)d_in[20], (const float*)d_in[21],
      (const float*)d_in[22], (const float*)d_in[23],
      out, B);
}
